// GCN_17463337026195
// MI455X (gfx1250) — matrix operand order repaired and checked by reading, not compiled, not run on hardware
//
#include <hip/hip_runtime.h>
#include <math.h>

// ---------------------------------------------------------------------------
// GCN forward on MI455X (gfx1250): 3x [ GEMM(WMMA bf16->f32) ; SpMM(atomics) ]
// then log_softmax. Wave32. GEMM uses 2x2 register blocking: each wave owns
// four 16x16 f32 accumulators fed by V_WMMA_F32_16X16X32_BF16, halving cache
// traffic per WMMA vs the 1x1 version.
// ---------------------------------------------------------------------------

#define N_NODES  50000
#define M_PAD    50016   // N_NODES padded to multiple of 32 (2 M-tiles/block)
#define N_EDGES  600000
#define NFEAT    256
#define NHID     128
#define NCLASS   40
#define NCLASS_P 64      // NCLASS padded to multiple of 32 for 2-tile waves

typedef __attribute__((ext_vector_type(16))) __bf16 v16bf;
typedef __attribute__((ext_vector_type(8)))  float  v8f;

union FragBF {          // 16 bf16 = 32 bytes = two 128-bit loads
    v16bf v;
    uint4 u[2];
};

static __device__ __forceinline__ unsigned short f32_to_bf16(float f) {
    unsigned int u = __float_as_uint(f);
    unsigned int rounding = 0x7FFFu + ((u >> 16) & 1u);   // round-to-nearest-even
    return (unsigned short)((u + rounding) >> 16);
}

// ---------------------------------------------------------------------------
// Elementwise casts
// ---------------------------------------------------------------------------
__global__ void cast_bf16_kernel(const float* __restrict__ in,
                                 unsigned short* __restrict__ out, int n) {
    int i = blockIdx.x * blockDim.x + threadIdx.x;
    if (i < n) out[i] = f32_to_bf16(in[i]);
}

__global__ void relu_cast_bf16_kernel(const float* __restrict__ in,
                                      unsigned short* __restrict__ out, int n) {
    int i = blockIdx.x * blockDim.x + threadIdx.x;
    if (i < n) out[i] = f32_to_bf16(fmaxf(in[i], 0.0f));
}

// W is KxN row-major f32; emit W^T as Npad x K bf16 (rows >= N zero-filled).
__global__ void transpose_cast_kernel(const float* __restrict__ W,
                                      unsigned short* __restrict__ WT,
                                      int K, int N, int Npad) {
    int i = blockIdx.x * blockDim.x + threadIdx.x;
    int total = Npad * K;
    if (i >= total) return;
    int n = i / K;
    int k = i - n * K;
    float v = (n < N) ? W[(size_t)k * N + n] : 0.0f;
    WT[(size_t)n * K + k] = f32_to_bf16(v);
}

// agg[node][f] = bias[f]
__global__ void fill_bias_kernel(float* __restrict__ agg,
                                 const float* __restrict__ b, int F, int n) {
    int i = blockIdx.x * blockDim.x + threadIdx.x;
    if (i < n) agg[i] = b[i % F];
}

// ---------------------------------------------------------------------------
// GEMM: C[M x N] = A[Mpad x K](bf16) * W[K x N]  (W given as WT: Npad x K bf16)
// Block = 32 rows x Npad cols; Npad/32 waves; wave w owns N-tiles {2w, 2w+1}
// and both M-tiles -> 2x2 accumulators, 4 WMMAs per 8 b128 loads per K-step.
// A fragment (16-bit A 16x32, ISA 7.12.2): lanes 0-15 row m+l, elems 0..7 ->
// K k0..k0+7, elems 8..15 -> K k0+16..k0+23; lanes 16-31 same row, K +8 / +24.
// B fragment (16-bit B 32x16): lane = column; same K order as A: lanes 0-15
// K k0..k0+7, k0+16..k0+23; lanes 16-31 K +8 / +24 (along W^T rows).
// ---------------------------------------------------------------------------
__global__ void __launch_bounds__(128)
gemm_wmma_bf16_2x2_kernel(const unsigned short* __restrict__ A,
                          const unsigned short* __restrict__ WT,
                          float* __restrict__ C,
                          int K, int N, int M) {
    const int wave = threadIdx.x >> 5;
    const int lane = threadIdx.x & 31;
    const int l15  = lane & 15;
    const int hi   = lane >> 4;               // 0: lanes 0-15, 1: lanes 16-31

    const int m0 = blockIdx.x << 5;           // first of two M-tiles
    const int m1 = m0 + 16;
    const int n0 = wave << 5;                 // first of two N-tiles

    const unsigned short* arow0 = A  + (size_t)(m0 + l15) * K + (hi ? 8 : 0);
    const unsigned short* arow1 = arow0 + (size_t)16 * K;
    const unsigned short* bcol0 = WT + (size_t)(n0 + l15) * K + (hi ? 8 : 0);
    const unsigned short* bcol1 = bcol0 + (size_t)16 * K;

    v8f acc00 = {}, acc01 = {}, acc10 = {}, acc11 = {};
    for (int k0 = 0; k0 < K; k0 += 32) {
        FragBF a0, a1, b0, b1;
        a0.u[0] = *(const uint4*)(arow0 + k0);
        a0.u[1] = *(const uint4*)(arow0 + k0 + 16);
        a1.u[0] = *(const uint4*)(arow1 + k0);
        a1.u[1] = *(const uint4*)(arow1 + k0 + 16);
        b0.u[0] = *(const uint4*)(bcol0 + k0);
        b0.u[1] = *(const uint4*)(bcol0 + k0 + 16);
        b1.u[0] = *(const uint4*)(bcol1 + k0);
        b1.u[1] = *(const uint4*)(bcol1 + k0 + 16);
        acc00 = __builtin_amdgcn_wmma_f32_16x16x32_bf16(false, a0.v, false, b0.v,
                                                        (short)0, acc00, false, false);
        acc01 = __builtin_amdgcn_wmma_f32_16x16x32_bf16(false, a0.v, false, b1.v,
                                                        (short)0, acc01, false, false);
        acc10 = __builtin_amdgcn_wmma_f32_16x16x32_bf16(false, a1.v, false, b0.v,
                                                        (short)0, acc10, false, false);
        acc11 = __builtin_amdgcn_wmma_f32_16x16x32_bf16(false, a1.v, false, b1.v,
                                                        (short)0, acc11, false, false);
    }

    // C/D layout: VGPR i -> row mTile + 8*hi + i, col nTile + l15
    const int col0 = n0 + l15;
    const int col1 = n0 + 16 + l15;
    float* c0 = C + (size_t)(m0 + (hi ? 8 : 0)) * N;
    if (col0 < N) {
#pragma unroll
        for (int i = 0; i < 8; ++i) c0[(size_t)i * N + col0] = acc00[i];
    }
    if (col1 < N) {
#pragma unroll
        for (int i = 0; i < 8; ++i) c0[(size_t)i * N + col1] = acc01[i];
    }
    if (m1 < M) {
        float* c1 = C + (size_t)(m1 + (hi ? 8 : 0)) * N;
        if (col0 < N) {
#pragma unroll
            for (int i = 0; i < 8; ++i) c1[(size_t)i * N + col0] = acc10[i];
        }
        if (col1 < N) {
#pragma unroll
            for (int i = 0; i < 8; ++i) c1[(size_t)i * N + col1] = acc11[i];
        }
    }
}

// ---------------------------------------------------------------------------
// SpMM scatter: agg[dst[e]] += w[e] * support[src[e]]  (per feature quad)
// support rows are L2-resident (25.6MB << 192MB L2); atomics resolve in L2.
// ---------------------------------------------------------------------------
__global__ void spmm_atomic_kernel(const float* __restrict__ support,
                                   const int* __restrict__ src,
                                   const int* __restrict__ dst,
                                   const float* __restrict__ ew,
                                   float* __restrict__ agg,
                                   int F, int quads, int nWork) {
    int idx = blockIdx.x * blockDim.x + threadIdx.x;
    if (idx >= nWork) return;
    int e = idx / quads;
    int q = idx - e * quads;
    int s = src[e];
    int d = dst[e];
    float w = ew[e];
    const float4 v = *(const float4*)(support + (size_t)s * F + (size_t)q * 4);
    float* out = agg + (size_t)d * F + (size_t)q * 4;
    atomicAdd(out + 0, v.x * w);
    atomicAdd(out + 1, v.y * w);
    atomicAdd(out + 2, v.z * w);
    atomicAdd(out + 3, v.w * w);
}

// ---------------------------------------------------------------------------
// Row-wise log_softmax over NCLASS entries
// ---------------------------------------------------------------------------
__global__ void log_softmax_kernel(const float* __restrict__ in,
                                   float* __restrict__ out, int rows, int C) {
    int r = blockIdx.x * blockDim.x + threadIdx.x;
    if (r >= rows) return;
    const float* x = in + (size_t)r * C;
    float m = -INFINITY;
    for (int c = 0; c < C; ++c) m = fmaxf(m, x[c]);
    float s = 0.0f;
    for (int c = 0; c < C; ++c) s += __expf(x[c] - m);
    float lse = m + __logf(s);
    float* o = out + (size_t)r * C;
    for (int c = 0; c < C; ++c) o[c] = x[c] - lse;
}

// ---------------------------------------------------------------------------
// Host-side orchestration
// ---------------------------------------------------------------------------
extern "C" void kernel_launch(void* const* d_in, const int* in_sizes, int n_in,
                              void* d_out, int out_size, void* d_ws, size_t ws_size,
                              hipStream_t stream) {
    const float* x   = (const float*)d_in[0];
    const float* ew  = (const float*)d_in[1];
    const float* W1  = (const float*)d_in[2];
    const float* b1  = (const float*)d_in[3];
    const float* W2  = (const float*)d_in[4];
    const float* b2  = (const float*)d_in[5];
    const float* W3  = (const float*)d_in[6];
    const float* b3  = (const float*)d_in[7];
    const int* esrc  = (const int*)d_in[8];
    const int* edst  = (const int*)d_in[9];
    float* out       = (float*)d_out;

    // workspace carve-up (all offsets 16B-aligned)
    char* ws = (char*)d_ws;
    size_t off = 0;
    unsigned short* bufA    = (unsigned short*)(ws + off); off += (size_t)M_PAD   * NFEAT * 2;   // 25.6MB
    float*          support = (float*)(ws + off);          off += (size_t)N_NODES * NHID  * 4;   // 25.6MB
    float*          agg     = (float*)(ws + off);          off += (size_t)N_NODES * NHID  * 4;   // 25.6MB
    unsigned short* WT1     = (unsigned short*)(ws + off); off += (size_t)NHID    * NFEAT * 2;
    unsigned short* WT2     = (unsigned short*)(ws + off); off += (size_t)NHID    * NHID  * 2;
    unsigned short* WT3     = (unsigned short*)(ws + off); off += (size_t)NCLASS_P* NHID  * 2;

    const int BT = 256;
    const int mBlocks = (N_NODES / 16 + 1) / 2;   // 1563 blocks of 32 rows

    // --- prep: casts + weight transposes -----------------------------------
    {
        int n = N_NODES * NFEAT;
        cast_bf16_kernel<<<(n + BT - 1) / BT, BT, 0, stream>>>(x, bufA, n);
    }
    {
        int n = NHID * NFEAT;
        transpose_cast_kernel<<<(n + BT - 1) / BT, BT, 0, stream>>>(W1, WT1, NFEAT, NHID, NHID);
        n = NHID * NHID;
        transpose_cast_kernel<<<(n + BT - 1) / BT, BT, 0, stream>>>(W2, WT2, NHID, NHID, NHID);
        n = NCLASS_P * NHID;
        transpose_cast_kernel<<<(n + BT - 1) / BT, BT, 0, stream>>>(W3, WT3, NHID, NCLASS, NCLASS_P);
    }

    // --- layer 1 ------------------------------------------------------------
    gemm_wmma_bf16_2x2_kernel<<<mBlocks, NHID, 0, stream>>>(bufA, WT1, support,
                                                            NFEAT, NHID, N_NODES);
    {
        int n = N_NODES * NHID;
        fill_bias_kernel<<<(n + BT - 1) / BT, BT, 0, stream>>>(agg, b1, NHID, n);
        int quads = NHID / 4;
        int nWork = N_EDGES * quads;
        spmm_atomic_kernel<<<(nWork + BT - 1) / BT, BT, 0, stream>>>(support, esrc, edst, ew, agg,
                                                                     NHID, quads, nWork);
        relu_cast_bf16_kernel<<<(n + BT - 1) / BT, BT, 0, stream>>>(agg, bufA, n);
    }

    // --- layer 2 ------------------------------------------------------------
    gemm_wmma_bf16_2x2_kernel<<<mBlocks, NHID, 0, stream>>>(bufA, WT2, support,
                                                            NHID, NHID, N_NODES);
    {
        int n = N_NODES * NHID;
        fill_bias_kernel<<<(n + BT - 1) / BT, BT, 0, stream>>>(agg, b2, NHID, n);
        int quads = NHID / 4;
        int nWork = N_EDGES * quads;
        spmm_atomic_kernel<<<(nWork + BT - 1) / BT, BT, 0, stream>>>(support, esrc, edst, ew, agg,
                                                                     NHID, quads, nWork);
        relu_cast_bf16_kernel<<<(n + BT - 1) / BT, BT, 0, stream>>>(agg, bufA, n);
    }

    // --- layer 3: N = 40 (Npad=64 -> 2 waves x 2 tiles; cols >= 40 masked) --
    gemm_wmma_bf16_2x2_kernel<<<mBlocks, NCLASS_P, 0, stream>>>(bufA, WT3, support,
                                                                NHID, NCLASS, N_NODES);
    {
        int n = N_NODES * NCLASS;
        fill_bias_kernel<<<(n + BT - 1) / BT, BT, 0, stream>>>(agg, b3, NCLASS, n);
        int quads = NCLASS / 4;
        int nWork = N_EDGES * quads;
        spmm_atomic_kernel<<<(nWork + BT - 1) / BT, BT, 0, stream>>>(support, esrc, edst, ew, agg,
                                                                     NCLASS, quads, nWork);
        log_softmax_kernel<<<(N_NODES + BT - 1) / BT, BT, 0, stream>>>(agg, out, N_NODES, NCLASS);
    }
}
